// EngramMemory_29540785062026
// MI455X (gfx1250) — matrix operand order repaired and checked by reading, not compiled, not run on hardware
//
#include <hip/hip_runtime.h>
#include <hip/hip_bf16.h>
#include <stdint.h>
#include <stddef.h>

// ---------------- problem constants ----------------
#define B_      4
#define S_      2048
#define D_      512
#define H_      4
#define T_      50000
#define HID_    1536
#define CONCAT_ 4096
#define NTOK    (B_ * S_)   // 8192

typedef __attribute__((ext_vector_type(16))) __bf16 v16bf;
typedef __attribute__((ext_vector_type(8)))  float  v8f;

struct Mults { unsigned long long m[8]; };

// ---------------- host: numpy RandomState(42) multipliers ----------------
// Legacy numpy randint(2, 2**31) with rng = 2^31-3 <= 0xFFFFFFFF uses 32-bit
// masked rejection: draw u32 & 0x7FFFFFFF, reject > 2^31-3, add off=2.
namespace {
struct MT19937 {
  uint32_t mt[624]; int mti;
  explicit MT19937(uint32_t s) {
    mt[0] = s;
    for (int i = 1; i < 624; i++)
      mt[i] = 1812433253u * (mt[i-1] ^ (mt[i-1] >> 30)) + (uint32_t)i;
    mti = 624;
  }
  uint32_t next() {
    if (mti >= 624) {
      for (int i = 0; i < 624; i++) {
        uint32_t y = (mt[i] & 0x80000000u) | (mt[(i+1) % 624] & 0x7fffffffu);
        uint32_t v = mt[(i+397) % 624] ^ (y >> 1);
        if (y & 1u) v ^= 2567483615u;
        mt[i] = v;
      }
      mti = 0;
    }
    uint32_t y = mt[mti++];
    y ^= y >> 11; y ^= (y << 7) & 2636928640u;
    y ^= (y << 15) & 4022730752u; y ^= y >> 18;
    return y;
  }
};
static void compute_mults(unsigned long long m[8]) {
  MT19937 g(42);
  for (int i = 0; i < 8; i++) {
    uint32_t v;
    do { v = g.next() & 0x7fffffffu; } while (v > 0x7ffffffdu);
    m[i] = 2ull + (unsigned long long)v;
  }
}
} // namespace

// ---------------- device helpers ----------------
__device__ __forceinline__ float block_reduce(float v, float* buf) {
  const int tid = threadIdx.x;
  buf[tid] = v;
  __syncthreads();
  #pragma unroll
  for (int s = 128; s > 0; s >>= 1) {
    if (tid < s) buf[tid] += buf[tid + s];
    __syncthreads();
  }
  float r = buf[0];
  __syncthreads();
  return r;
}

// CDNA5 async global->LDS copy (no VGPR staging), tracked by ASYNCcnt.
__device__ __forceinline__ void async_copy_b128(unsigned lds_off, const void* gptr) {
  asm volatile("global_load_async_to_lds_b128 %0, %1, off"
               :: "v"(lds_off), "v"((unsigned long long)(uintptr_t)gptr)
               : "memory");
}
__device__ __forceinline__ void wait_asynccnt0() {
  asm volatile("s_wait_asynccnt 0x0" ::: "memory");
}

// ---------------- kernel: f32 -> bf16 weight conversion ----------------
__global__ void cvt_f32_bf16_kernel(const float* __restrict__ in,
                                    __hip_bfloat16* __restrict__ out, int n) {
  int i = blockIdx.x * blockDim.x + threadIdx.x;
  const int stride = gridDim.x * blockDim.x;
  for (; i < n; i += stride) out[i] = __float2bfloat16(in[i]);
}

// ---------------- kernel: hash + gather + LayerNorm(4096) -> bf16 ----------------
__global__ __launch_bounds__(256) void retrieve_ln_kernel(
    const int* __restrict__ tok, const float* __restrict__ tables,
    const float* __restrict__ rn_w, const float* __restrict__ rn_b,
    __hip_bfloat16* __restrict__ out, Mults mm) {
  __shared__ int   hsh[8];
  __shared__ float red[256];
  const int token = blockIdx.x;
  const int b = token / S_;
  const int s = token % S_;
  const int tid = threadIdx.x;

  if (tid < 8) {
    const int o = tid >> 2;          // 0 -> order 2, 1 -> order 3
    const int n = o ? 3 : 2;
    const unsigned long long mlt = mm.m[tid];
    unsigned long long h = 0;
    for (int j = 0; j < n; j++) {
      const int idx = s - (n - 1) + j;
      const unsigned long long t =
          (idx >= 0) ? (unsigned long long)(unsigned)tok[b * S_ + idx] : 0ull;
      h = (h * mlt + t) % (unsigned long long)T_;
    }
    hsh[tid] = (int)h;
  }
  __syncthreads();

  float x[16];
  float sum = 0.f, sq = 0.f;
  #pragma unroll
  for (int i = 0; i < 16; i++) {
    const int c   = tid + i * 256;       // 0..4095
    const int tab = c >> 9;              // table index 0..7
    const int col = c & 511;
    const int nm1 = (tab < 4) ? 1 : 2;   // order-1 mask threshold
    float v = 0.f;
    if (s >= nm1)
      v = tables[((size_t)tab * T_ + (size_t)hsh[tab]) * D_ + col];
    x[i] = v; sum += v; sq += v * v;
  }
  sum = block_reduce(sum, red);
  sq  = block_reduce(sq, red);
  const float mean = sum * (1.f / CONCAT_);
  const float var  = sq * (1.f / CONCAT_) - mean * mean;
  const float rstd = rsqrtf(var + 1e-5f);
  #pragma unroll
  for (int i = 0; i < 16; i++) {
    const int c = tid + i * 256;
    const float y = (x[i] - mean) * rstd * rn_w[c] + rn_b[c];
    out[(size_t)token * CONCAT_ + c] = __float2bfloat16(y);
  }
}

// ---------------- kernel: LayerNorm(1536) on hidden -> bf16 ----------------
__global__ __launch_bounds__(256) void hidden_ln_kernel(
    const float* __restrict__ hs, const float* __restrict__ w,
    const float* __restrict__ b, __hip_bfloat16* __restrict__ out) {
  __shared__ float red[256];
  const int token = blockIdx.x;
  const int tid = threadIdx.x;
  float x[6];
  float sum = 0.f, sq = 0.f;
  #pragma unroll
  for (int i = 0; i < 6; i++) {
    const int c = tid + i * 256;
    const float v = hs[(size_t)token * HID_ + c];
    x[i] = v; sum += v; sq += v * v;
  }
  sum = block_reduce(sum, red);
  sq  = block_reduce(sq, red);
  const float mean = sum * (1.f / HID_);
  const float var  = sq * (1.f / HID_) - mean * mean;
  const float rstd = rsqrtf(var + 1e-5f);
  #pragma unroll
  for (int i = 0; i < 6; i++) {
    const int c = tid + i * 256;
    out[(size_t)token * HID_ + c] =
        __float2bfloat16((x[i] - mean) * rstd * w[c] + b[c]);
  }
}

// ---------------- kernel: tiled bf16 WMMA GEMM  C[M,N] = A[M,K] * W[N,K]^T + bias ----------------
// 256 threads = 8 waves; block tile 128x64, BK=64, double-buffered LDS filled by
// global_load_async_to_lds_b128; wave -> 32x32 output = 2x2 WMMA frags x 2 k-steps.
__global__ __launch_bounds__(256) void gemm_bf16_nt(
    const __hip_bfloat16* __restrict__ A, const __hip_bfloat16* __restrict__ W,
    const float* __restrict__ bias, float* __restrict__ C,
    int M, int N, int K) {
  constexpr int BM = 128, BN = 64, BK = 64;
  __shared__ __align__(16) __hip_bfloat16 As[2][BM][BK];  // 32 KB
  __shared__ __align__(16) __hip_bfloat16 Ws[2][BN][BK];  // 16 KB

  const int tid  = threadIdx.x;
  const int lane = tid & 31;
  const int wave = tid >> 5;
  const int wm   = wave >> 1;   // 0..3: 32-row slab
  const int wn   = wave & 1;    // 0..1: 32-col slab
  const int l15  = lane & 15;
  const int lh   = lane >> 4;

  const int bm0 = blockIdx.x * BM;
  const int bn0 = blockIdx.y * BN;

  const unsigned as_base[2] = {(unsigned)(uintptr_t)&As[0][0][0],
                               (unsigned)(uintptr_t)&As[1][0][0]};
  const unsigned ws_base[2] = {(unsigned)(uintptr_t)&Ws[0][0][0],
                               (unsigned)(uintptr_t)&Ws[1][0][0]};

  // stage tiles for K offset k0 into LDS buffer `buf` (async, no VGPR staging)
  auto stage = [&](int k0, int buf) {
    // A tile: 128x64 bf16 = 1024 uint4, 4 per thread
    #pragma unroll
    for (int t = 0; t < 4; t++) {
      const int idx = tid + t * 256;
      const int row = idx >> 3;
      const int seg = idx & 7;
      async_copy_b128(as_base[buf] + (unsigned)((row * BK + seg * 8) * 2),
                      A + (size_t)(bm0 + row) * K + k0 + seg * 8);
    }
    // W tile: 64x64 bf16 = 512 uint4, 2 per thread
    #pragma unroll
    for (int t = 0; t < 2; t++) {
      const int idx = tid + t * 256;
      const int row = idx >> 3;
      const int seg = idx & 7;
      async_copy_b128(ws_base[buf] + (unsigned)((row * BK + seg * 8) * 2),
                      W + (size_t)(bn0 + row) * K + k0 + seg * 8);
    }
  };

  v8f acc[2][2] = {};
  union Frag { uint4 u[2]; v16bf v; };

  int cur = 0;
  stage(0, 0);
  wait_asynccnt0();
  __syncthreads();

  for (int k0 = 0; k0 < K; k0 += BK) {
    if (k0 + BK < K) stage(k0 + BK, cur ^ 1);

    // fragment row base pointers in current buffer (uint4 stride: 8 per row)
    const uint4* ap0 = (const uint4*)&As[cur][wm * 32 + l15][0];
    const uint4* ap1 = (const uint4*)&As[cur][wm * 32 + 16 + l15][0];
    const uint4* bp0 = (const uint4*)&Ws[cur][wn * 32 + l15][0];
    const uint4* bp1 = (const uint4*)&Ws[cur][wn * 32 + 16 + l15][0];

    #pragma unroll
    for (int kk = 0; kk < 2; kk++) {           // two 16x16x32 k-steps per BK=64
      Frag a[2], b[2];
      // A frag (16x32): lane -> M = l&15; K runs {lh*8..+7} and {16+lh*8..+7}
      a[0].u[0] = ap0[kk * 4 + lh];     a[0].u[1] = ap0[kk * 4 + 2 + lh];
      a[1].u[0] = ap1[kk * 4 + lh];     a[1].u[1] = ap1[kk * 4 + 2 + lh];
      // B frag (32x16): lane -> N = l&15; K runs {lh*8..+7} and {16+lh*8..+7}
      b[0].u[0] = bp0[kk * 4 + lh];     b[0].u[1] = bp0[kk * 4 + 2 + lh];
      b[1].u[0] = bp1[kk * 4 + lh];     b[1].u[1] = bp1[kk * 4 + 2 + lh];
      #pragma unroll
      for (int ti = 0; ti < 2; ti++)
        #pragma unroll
        for (int tj = 0; tj < 2; tj++)
          acc[ti][tj] = __builtin_amdgcn_wmma_f32_16x16x32_bf16(
              false, a[ti].v, false, b[tj].v, (short)0, acc[ti][tj], false, false);
    }

    wait_asynccnt0();   // next tile resident before anyone flips buffers
    __syncthreads();    // codegen drains DScnt before barrier -> safe to overwrite
    cur ^= 1;
  }

  // Epilogue: C/D layout — VGPR r: lanes 0-15 -> M=r, lanes 16-31 -> M=8+r; N = l&15.
  #pragma unroll
  for (int ti = 0; ti < 2; ti++) {
    #pragma unroll
    for (int tj = 0; tj < 2; tj++) {
      const int m0 = bm0 + wm * 32 + ti * 16 + lh * 8;
      const int n  = bn0 + wn * 32 + tj * 16 + l15;
      const float bv = bias[n];
      #pragma unroll
      for (int r = 0; r < 8; r++)
        C[(size_t)(m0 + r) * N + n] = acc[ti][tj][r] + bv;
    }
  }
}

// ---------------- kernel: gate = sigmoid(q·k/sqrt(D)), gated = gate*v (bf16) ----------------
__global__ __launch_bounds__(256) void gate_kernel(
    const float* __restrict__ q, const float* __restrict__ k,
    const float* __restrict__ v, float* __restrict__ gate_out,
    __hip_bfloat16* __restrict__ gated) {
  __shared__ float red[256];
  const int token = blockIdx.x;
  const int tid = threadIdx.x;
  float part = 0.f;
  #pragma unroll
  for (int i = 0; i < 2; i++) {
    const size_t off = (size_t)token * D_ + tid + i * 256;
    part += q[off] * k[off];
  }
  const float dot = block_reduce(part, red);
  const float g = 1.f / (1.f + __expf(-dot * 0.044194173824159216f)); // 1/sqrt(512)
  if (tid == 0) gate_out[token] = g;
  #pragma unroll
  for (int i = 0; i < 2; i++) {
    const size_t off = (size_t)token * D_ + tid + i * 256;
    gated[off] = __float2bfloat16(g * v[off]);
  }
}

// ---------------- kernel: final LayerNorm(512) -> f32 out ----------------
__global__ __launch_bounds__(256) void final_ln_kernel(
    const float* __restrict__ in, const float* __restrict__ w,
    const float* __restrict__ b, float* __restrict__ out) {
  __shared__ float red[256];
  const int token = blockIdx.x;
  const int tid = threadIdx.x;
  float x[2];
  float sum = 0.f, sq = 0.f;
  #pragma unroll
  for (int i = 0; i < 2; i++) {
    const float v = in[(size_t)token * D_ + tid + i * 256];
    x[i] = v; sum += v; sq += v * v;
  }
  sum = block_reduce(sum, red);
  sq  = block_reduce(sq, red);
  const float mean = sum * (1.f / D_);
  const float var  = sq * (1.f / D_) - mean * mean;
  const float rstd = rsqrtf(var + 1e-5f);
  #pragma unroll
  for (int i = 0; i < 2; i++) {
    const int c = tid + i * 256;
    out[(size_t)token * D_ + c] = (x[i] - mean) * rstd * w[c] + b[c];
  }
}

// ---------------- workspace layout (bytes) ----------------
#define OFF_RN   ((size_t)0)                        // bf16 [8192,4096]  64 MB
#define OFF_HN   ((size_t)67108864)                 // bf16 [8192,1536]  24 MB
#define OFF_Q    ((size_t)92274688)                 // f32  [8192,512]   16 MB
#define OFF_K    ((size_t)109051904)                // f32  [8192,512]   16 MB
#define OFF_V    ((size_t)125829120)                // f32  [8192,512]   16 MB
#define OFF_WQ   ((size_t)142606336)                // bf16 [512,1536]
#define OFF_WK   ((size_t)144179200)                // bf16 [512,4096]
#define OFF_WV   ((size_t)148373504)                // bf16 [512,4096]
#define OFF_WO   ((size_t)152567808)                // bf16 [512,512]
#define OFF_GATED OFF_HN                            // bf16 [8192,512] (hn dead)
#define OFF_PROJ  OFF_RN                            // f32  [8192,512] (rn dead)

extern "C" void kernel_launch(void* const* d_in, const int* in_sizes, int n_in,
                              void* d_out, int out_size, void* d_ws, size_t ws_size,
                              hipStream_t stream) {
  const int*   tok    = (const int*)d_in[0];
  const float* hidden = (const float*)d_in[1];
  const float* tables = (const float*)d_in[2];
  const float* gq_w = (const float*)d_in[3];
  const float* gq_b = (const float*)d_in[4];
  const float* gk_w = (const float*)d_in[5];
  const float* gk_b = (const float*)d_in[6];
  const float* gv_w = (const float*)d_in[7];
  const float* gv_b = (const float*)d_in[8];
  const float* rn_w = (const float*)d_in[9];
  const float* rn_b = (const float*)d_in[10];
  const float* qn_w = (const float*)d_in[11];
  const float* qn_b = (const float*)d_in[12];
  const float* op_w = (const float*)d_in[13];
  const float* op_b = (const float*)d_in[14];
  const float* on_w = (const float*)d_in[15];
  const float* on_b = (const float*)d_in[16];

  char* ws = (char*)d_ws;
  __hip_bfloat16* rn    = (__hip_bfloat16*)(ws + OFF_RN);
  __hip_bfloat16* hn    = (__hip_bfloat16*)(ws + OFF_HN);
  float*          q     = (float*)(ws + OFF_Q);
  float*          k     = (float*)(ws + OFF_K);
  float*          v     = (float*)(ws + OFF_V);
  __hip_bfloat16* wq    = (__hip_bfloat16*)(ws + OFF_WQ);
  __hip_bfloat16* wk    = (__hip_bfloat16*)(ws + OFF_WK);
  __hip_bfloat16* wv    = (__hip_bfloat16*)(ws + OFF_WV);
  __hip_bfloat16* wo    = (__hip_bfloat16*)(ws + OFF_WO);
  __hip_bfloat16* gated = (__hip_bfloat16*)(ws + OFF_GATED);
  float*          proj  = (float*)(ws + OFF_PROJ);

  Mults mm;
  compute_mults(mm.m);

  // Weight conversions f32 -> bf16
  cvt_f32_bf16_kernel<<<1024, 256, 0, stream>>>(gq_w, wq, D_ * HID_);
  cvt_f32_bf16_kernel<<<2048, 256, 0, stream>>>(gk_w, wk, D_ * CONCAT_);
  cvt_f32_bf16_kernel<<<2048, 256, 0, stream>>>(gv_w, wv, D_ * CONCAT_);
  cvt_f32_bf16_kernel<<<512,  256, 0, stream>>>(op_w, wo, D_ * D_);

  // Fused retrieve + LayerNorm, and hidden LayerNorm
  retrieve_ln_kernel<<<NTOK, 256, 0, stream>>>(tok, tables, rn_w, rn_b, rn, mm);
  hidden_ln_kernel<<<NTOK, 256, 0, stream>>>(hidden, qn_w, qn_b, hn);

  // q/k/v projections via WMMA GEMM
  dim3 grid(NTOK / 128, D_ / 64);
  gemm_bf16_nt<<<grid, 256, 0, stream>>>(hn, wq, gq_b, q, NTOK, D_, HID_);
  gemm_bf16_nt<<<grid, 256, 0, stream>>>(rn, wk, gk_b, k, NTOK, D_, CONCAT_);
  gemm_bf16_nt<<<grid, 256, 0, stream>>>(rn, wv, gv_b, v, NTOK, D_, CONCAT_);

  // Gate + gated values
  float* gate_out = (float*)d_out + (size_t)NTOK * D_;
  gate_kernel<<<NTOK, 256, 0, stream>>>(q, k, v, gate_out, gated);

  // Output projection + final LayerNorm
  gemm_bf16_nt<<<grid, 256, 0, stream>>>(gated, wo, op_b, proj, NTOK, D_, D_);
  final_ln_kernel<<<NTOK, 256, 0, stream>>>(proj, on_w, on_b, (float*)d_out);
}
